// LSTMGC_22978075033634
// MI455X (gfx1250) — matrix operand order repaired and checked by reading, not compiled, not run on hardware
//
#include <hip/hip_runtime.h>

// ---------------- problem constants ----------------
#define N_NODES 5000
#define N_EDGES 80000
#define BATCH   64
#define T_SEQ   12
#define UNITS   64
#define GATES   256           // 4*UNITS
#define NB      (N_NODES*BATCH)   // 320000 sequences

typedef __attribute__((ext_vector_type(16))) _Float16 v16h;
typedef __attribute__((ext_vector_type(8)))  float    v8f;

union FragAB { v16h v; unsigned u[8]; _Float16 h[16]; };

// ---------------- workspace layout (dwords) ----------------
static constexpr long S1_OFF  = 0;
static constexpr long S2_OFF  = (long)T_SEQ * NB;          // 3,840,000
static constexpr long UP_OFF  = 2L * T_SEQ * NB;           // 7,680,000
static constexpr long WX1_OFF = UP_OFF + 8192;
static constexpr long WX2_OFF = WX1_OFF + 256;
static constexpr long BI_OFF  = WX2_OFF + 256;
static constexpr long CNT_OFF = BI_OFF + 256;

// ---------------- kernel 1: zero s2 + counts ----------------
__global__ void k_zero(float* __restrict__ s2, int* __restrict__ cnt) {
    int i = blockIdx.x * blockDim.x + threadIdx.x;
    if (i < T_SEQ * NB) s2[i] = 0.0f;
    if (i < N_NODES)    cnt[i] = 0;
}

// ---------------- kernel 2: edge in-degree counts ----------------
__global__ void k_count(const int* __restrict__ src, int* __restrict__ cnt) {
    int e = blockIdx.x * blockDim.x + threadIdx.x;
    if (e < N_EDGES) atomicAdd(&cnt[src[e]], 1);
}

// ---------------- kernel 3: edge scatter (segment sum) ----------------
// s2[t*NB + src*B + b] += x[b,t,dst]
__global__ void k_scatter(const float* __restrict__ x,
                          const int* __restrict__ src, const int* __restrict__ dst,
                          float* __restrict__ s2) {
    int e = blockIdx.x;
    int s = src[e], d = dst[e];
    for (int bt = threadIdx.x; bt < BATCH * T_SEQ; bt += blockDim.x) {
        int b = bt / T_SEQ, t = bt - b * T_SEQ;       // bt = b*T + t
        float val = x[(long)bt * N_NODES + d];
        atomicAdd(&s2[(long)t * NB + s * BATCH + b], val);
    }
}

// ---------------- kernel 4: mean-divide + transpose x into s1 ----------------
__global__ void k_finalize(const float* __restrict__ x, const int* __restrict__ cnt,
                           float* __restrict__ s1, float* __restrict__ s2) {
    long i = (long)blockIdx.x * blockDim.x + threadIdx.x;
    if (i >= (long)T_SEQ * NB) return;
    int t   = (int)(i / NB);
    int rem = (int)(i - (long)t * NB);
    int n = rem >> 6;            // /BATCH (=64)
    int b = rem & 63;
    int c = cnt[n]; if (c < 1) c = 1;
    s2[i] = s2[i] / (float)c;
    s1[i] = x[((long)b * T_SEQ + t) * N_NODES + n];
}

// ---------------- kernel 5: pack U to WMMA-B frag order; fold gc_w into wx1/wx2 ----
__global__ void k_pack(const float* __restrict__ lstm_U, const float* __restrict__ gc_w,
                       const float* __restrict__ lstm_W, const float* __restrict__ lstm_b,
                       unsigned* __restrict__ upack, float* __restrict__ wx1,
                       float* __restrict__ wx2, float* __restrict__ bias) {
    int id = blockIdx.x * blockDim.x + threadIdx.x;
    if (id < 8192) {
        int v    = id & 7;
        int lane = (id >> 3) & 31;
        int kc   = (id >> 8) & 1;
        int nt   = id >> 9;
        int n = nt * 16 + (lane & 15);
        int K = kc * 32 + ((v < 4) ? 2 * v : 16 + 2 * (v - 4)) + (lane >> 4) * 8;
        union { _Float16 h[2]; unsigned u; } p;
        p.h[0] = (_Float16)lstm_U[(long)K * GATES + n];
        p.h[1] = (_Float16)lstm_U[(long)(K + 1) * GATES + n];
        upack[id] = p.u;
    }
    if (id < GATES) {
        float a = 0.0f, c = 0.0f;
        #pragma unroll
        for (int j = 0; j < 10; ++j) {
            a += gc_w[j] * lstm_W[j * GATES + id];
            c += gc_w[j] * lstm_W[(10 + j) * GATES + id];
        }
        wx1[id] = a; wx2[id] = c; bias[id] = lstm_b[id];
    }
}

// ---------------- kernel 6: fused LSTM (WMMA) + dense head ----------------
__launch_bounds__(128, 1)
__global__ void k_lstm(const float* __restrict__ s1g, const float* __restrict__ s2g,
                       const unsigned* __restrict__ upack,
                       const float* __restrict__ wx1g, const float* __restrict__ wx2g,
                       const float* __restrict__ biasg,
                       const float* __restrict__ dW, const float* __restrict__ db,
                       float* __restrict__ out) {
    __shared__ unsigned  Ulds[8192];       // 32 KB: B fragments
    __shared__ float     wx1[GATES], wx2[GATES], bias[GATES];
    __shared__ _Float16  hlds[64 * UNITS]; // 8 KB: hidden state (f16, row-major)
    __shared__ float     s1l[4][16], s2l[4][16];
    __shared__ float     dWl[UNITS * 3];
    __shared__ float     dbl[3];

    const int tid  = threadIdx.x;
    const int lane = tid & 31;
    const int wave = tid >> 5;
    const int nl   = lane & 15;
    const int grp  = lane >> 4;
    const int node = blockIdx.x;
    const long rowBase = (long)node * 64;

    for (int i = tid; i < 8192; i += 128) Ulds[i] = upack[i];
    for (int i = tid; i < GATES; i += 128) { wx1[i] = wx1g[i]; wx2[i] = wx2g[i]; bias[i] = biasg[i]; }
    for (int i = tid; i < 64 * UNITS; i += 128) hlds[i] = (_Float16)0.0f;
    for (int i = tid; i < UNITS * 3; i += 128) dWl[i] = dW[i];
    if (tid < 3) dbl[tid] = db[tid];

    float creg[32];                       // c state: creg[q*8+v], rows m=8*grp+v, u=q*16+nl
    #pragma unroll
    for (int i = 0; i < 32; ++i) creg[i] = 0.0f;

    __syncthreads();

    for (int t = 0; t < T_SEQ; ++t) {
        if (lane < 16)      s1l[wave][lane]      = s1g[(long)t * NB + rowBase + wave * 16 + lane];
        else                s2l[wave][lane - 16] = s2g[(long)t * NB + rowBase + wave * 16 + (lane - 16)];
        if (t + 1 < T_SEQ && lane < 16) {
            __builtin_prefetch(&s1g[(long)(t + 1) * NB + rowBase + wave * 16 + lane], 0, 1);
            __builtin_prefetch(&s2g[(long)(t + 1) * NB + rowBase + wave * 16 + lane], 0, 1);
        }
        __syncthreads();

        // A fragments: row = wave*16+nl; K striping per ISA 16-bit A layout
        FragAB a0, a1;
        #pragma unroll
        for (int v = 0; v < 8; ++v) {
            int kbase = ((v < 4) ? 2 * v : 16 + 2 * (v - 4)) + grp * 8;
            const unsigned* hp = reinterpret_cast<const unsigned*>(&hlds[(wave * 16 + nl) * UNITS + kbase]);
            a0.u[v] = hp[0];
            a1.u[v] = hp[16];              // + 32 halves = K-chunk 1
        }
        float s1r[8], s2r[8];
        #pragma unroll
        for (int v = 0; v < 8; ++v) { int m = grp * 8 + v; s1r[v] = s1l[wave][m]; s2r[v] = s2l[wave][m]; }

        v8f acc[16];
        #pragma unroll
        for (int nt = 0; nt < 16; ++nt) {
            int n = nt * 16 + nl;
            float w1 = wx1[n], w2 = wx2[n], bb = bias[n];
            #pragma unroll
            for (int v = 0; v < 8; ++v)
                acc[nt][v] = fmaf(s1r[v], w1, fmaf(s2r[v], w2, bb));   // folded input GEMM
            FragAB b0, b1;
            #pragma unroll
            for (int v = 0; v < 8; ++v) {
                b0.u[v] = Ulds[((nt * 2 + 0) * 32 + lane) * 8 + v];
                b1.u[v] = Ulds[((nt * 2 + 1) * 32 + lane) * 8 + v];
            }
            acc[nt] = __builtin_amdgcn_wmma_f32_16x16x32_f16(false, a0.v, false, b0.v,
                                                             (short)0, acc[nt], false, false);
            acc[nt] = __builtin_amdgcn_wmma_f32_16x16x32_f16(false, a1.v, false, b1.v,
                                                             (short)0, acc[nt], false, false);
        }

        // gate math: tile g*4+q holds gate g, unit u=q*16+nl, row m=8*grp+v
        #pragma unroll
        for (int q = 0; q < 4; ++q) {
            #pragma unroll
            for (int v = 0; v < 8; ++v) {
                float zi = acc[0 * 4 + q][v];
                float zf = acc[1 * 4 + q][v];
                float zg = acc[2 * 4 + q][v];
                float zo = acc[3 * 4 + q][v];
                float ig = 1.0f / (1.0f + __expf(-zi));
                float fg = 1.0f / (1.0f + __expf(-zf));
                float og = 1.0f / (1.0f + __expf(-zo));
                float gg = fmaxf(zg, 0.0f);
                float c  = fg * creg[q * 8 + v] + ig * gg;
                creg[q * 8 + v] = c;
                float h  = og * fmaxf(c, 0.0f);
                hlds[(wave * 16 + grp * 8 + v) * UNITS + q * 16 + nl] = (_Float16)h;
            }
        }
        __syncthreads();
    }

    // dense head: out[b, j, n] = hT[row] . dW[:,j] + db[j];  local row == b, n == node
    for (int o = tid; o < 64 * 3; o += 128) {
        int row = o / 3, j = o - row * 3;
        float sum = dbl[j];
        #pragma unroll 8
        for (int u = 0; u < UNITS; ++u)
            sum += (float)hlds[row * UNITS + u] * dWl[u * 3 + j];
        out[((long)row * 3 + j) * N_NODES + node] = sum;
    }
}

// ---------------- launcher ----------------
extern "C" void kernel_launch(void* const* d_in, const int* in_sizes, int n_in,
                              void* d_out, int out_size, void* d_ws, size_t ws_size,
                              hipStream_t stream) {
    const float* x       = (const float*)d_in[0];
    const float* gc_w    = (const float*)d_in[1];
    const float* lstm_W  = (const float*)d_in[2];
    const float* lstm_U  = (const float*)d_in[3];
    const float* lstm_b  = (const float*)d_in[4];
    const float* dense_W = (const float*)d_in[5];
    const float* dense_b = (const float*)d_in[6];
    const int*   e_src   = (const int*)d_in[7];
    const int*   e_dst   = (const int*)d_in[8];

    float*    wsf   = (float*)d_ws;
    float*    s1    = wsf + S1_OFF;
    float*    s2    = wsf + S2_OFF;
    unsigned* upack = (unsigned*)(wsf + UP_OFF);
    float*    wx1   = wsf + WX1_OFF;
    float*    wx2   = wsf + WX2_OFF;
    float*    bias  = wsf + BI_OFF;
    int*      cnt   = (int*)(wsf + CNT_OFF);
    float*    out   = (float*)d_out;

    const long tot = (long)T_SEQ * NB;
    k_zero    <<<(int)((tot + 255) / 256), 256, 0, stream>>>(s2, cnt);
    k_count   <<<(N_EDGES + 255) / 256,    256, 0, stream>>>(e_src, cnt);
    k_scatter <<<N_EDGES,                  256, 0, stream>>>(x, e_src, e_dst, s2);
    k_finalize<<<(int)((tot + 255) / 256), 256, 0, stream>>>(x, cnt, s1, s2);
    k_pack    <<<(8192 + 255) / 256,       256, 0, stream>>>(lstm_U, gc_w, lstm_W, lstm_b,
                                                             upack, wx1, wx2, bias);
    k_lstm    <<<N_NODES,                  128, 0, stream>>>(s1, s2, upack, wx1, wx2, bias,
                                                             dense_W, dense_b, out);
}
